// LSTM_41721312313406
// MI455X (gfx1250) — matrix operand order repaired and checked by reading, not compiled, not run on hardware
//
#include <hip/hip_runtime.h>
#include <hip/hip_bf16.h>

// ---------------------------------------------------------------------------
// LSTM on MI455X (gfx1250, wave32, WMMA).
//   B=128, T=256, H=1024, C=10, IN=1
// Strategy: persistent kernel, 64 workgroups x 256 threads (8 waves).
//  - Gate GEMMs in bf16 via v_wmma_f32_16x16x32_bf16 (fp32 accum).
//  - Per-WG weight slice (128 KB) staged ONCE into LDS (320 KB/WGP);
//    steady-state B fragments come from ds_load_b128 (step-invariant).
//  - h double-buffered in global bf16; c kept entirely in registers.
//  - Device-wide spin barrier (atomic + agent fences) between timesteps.
//  - Branch-free activations on the trans pipe (v_exp_f32 / v_rcp_f32 /
//    v_tanh_f32) to keep the serial per-step critical path short.
// ---------------------------------------------------------------------------

#define BSZ 128
#define TSZ 256
#define HSZ 1024
#define CSZ 10
#define NWG 64     // workgroups; each owns 16 hidden columns (x 4 gates)
#define KT  32     // K tiles: 1024 / 32

typedef __attribute__((ext_vector_type(16))) __bf16 v16bf;
typedef __attribute__((ext_vector_type(8)))  __bf16 v8bf;
typedef __attribute__((ext_vector_type(8)))  float  v8f;

union AFrag { v16bf v; v8bf h[2]; };

#define LOG2E 1.4426950408889634f

__device__ __forceinline__ float fast_exp(float x) {
    return __builtin_amdgcn_exp2f(x * LOG2E);          // v_exp_f32 (base-2)
}
__device__ __forceinline__ float fast_sigmoid(float x) {
    // Saturates correctly: x->-inf: exp2->+inf -> rcp -> 0 ; x->+inf: -> 1.
    return __builtin_amdgcn_rcpf(1.0f + fast_exp(-x)); // v_rcp_f32
}
__device__ __forceinline__ float fast_tanh(float x) {
#if __has_builtin(__builtin_amdgcn_tanhf)
    return __builtin_amdgcn_tanhf(x);                  // v_tanh_f32
#else
    return 2.0f * fast_sigmoid(2.0f * x) - 1.0f;       // branch-free fallback
#endif
}

// ---------------------------------------------------------------------------
// Pack W (HxH row-major fp32, W[k][col]) for 4 gates into bf16 B-fragment
// blocks. Block = (gate G, col-tile n, K-tile kt): 32x16 bf16, stored so that
// lane L reads 16 contiguous bf16 at (block*512 + L*16).
// B layout used here (32x16, 16-bit, contiguous k, same as A): N = L%16 ; K = (L/16)*16 + e, e=0..15.
// ---------------------------------------------------------------------------
__global__ void lstm_pack_w(const float* __restrict__ Wg, const float* __restrict__ Wi,
                            const float* __restrict__ Wf, const float* __restrict__ Wo,
                            __bf16* __restrict__ packed)
{
    // id = (((G*NWG + n)*KT + kt)*32 + L)*16 + e   ; total 4*2^20 elements
    size_t id = (size_t)blockIdx.x * blockDim.x + threadIdx.x;
    int e  = (int)(id & 15);
    int L  = (int)((id >> 4)  & 31);
    int kt = (int)((id >> 9)  & 31);
    int n  = (int)((id >> 14) & 63);
    int G  = (int)(id >> 20);
    const float* W = (G == 0) ? Wg : (G == 1) ? Wi : (G == 2) ? Wf : Wo;
    int k   = kt * 32 + (L >> 4) * 16 + e;
    int col = n * 16 + (L & 15);
    packed[id] = (__bf16)W[(size_t)k * HSZ + col];
}

// ---------------------------------------------------------------------------
// Persistent timestep kernel.
// Dynamic LDS: 4 gates * KT * 512 bf16 = 131072 B (fits 320 KB WGP LDS).
// ---------------------------------------------------------------------------
__global__ void __launch_bounds__(256, 1)
lstm_persistent(const float* __restrict__ x,
                const float* __restrict__ Wgx, const float* __restrict__ Bg,
                const float* __restrict__ Wix, const float* __restrict__ Bi,
                const float* __restrict__ Wfx, const float* __restrict__ Bf,
                const float* __restrict__ Wox, const float* __restrict__ Bo,
                const __bf16* __restrict__ packedW,
                __bf16* __restrict__ hbuf,          // 2 x (128*1024) bf16
                unsigned* __restrict__ barrier_ctr)
{
    extern __shared__ __bf16 lds_w[];               // [4][KT][512]

    const int wg   = blockIdx.x;            // 0..63 : hidden cols [wg*16, wg*16+16)
    const int w    = threadIdx.x >> 5;      // wave 0..7 : batch rows [w*16, w*16+16)
    const int lane = threadIdx.x & 31;
    const int half = lane >> 4;
    const int nlan = lane & 15;
    const int col  = wg * 16 + nlan;        // hidden column owned by this lane (N)
    const int aRow = w * 16 + nlan;         // batch row for A-fragment (M = lane%16)

    // ---- one-time: stage this WG's weight slice into LDS (128 KB) ----
    {
        const int chunksPerGate = (KT * 512) / 8;   // v8bf (16 B) chunks = 2048
        for (int G = 0; G < 4; ++G) {
            const v8bf* src = reinterpret_cast<const v8bf*>(
                packedW + ((size_t)(G * NWG + wg) * KT) * 512);
            v8bf* dst = reinterpret_cast<v8bf*>(lds_w + (size_t)G * KT * 512);
            for (int i = threadIdx.x; i < chunksPerGate; i += 256)
                dst[i] = src[i];
        }
    }
    __syncthreads();

    // Per-lane step-invariant scalars.
    const float wx_g = Wgx[col], wx_i = Wix[col], wx_f = Wfx[col], wx_o = Wox[col];
    const float bb_g = Bg[col],  bb_i = Bi[col],  bb_f = Bf[col],  bb_o = Bo[col];

    // Cell state lives in registers: lane owns (b = w*16 + half*8 + r, col).
    float creg[8];
#pragma unroll
    for (int r = 0; r < 8; ++r) creg[r] = 0.0f;

    const int lbo = lane * 16;                      // lane offset within a fragment block

    for (int t = 0; t < TSZ; ++t) {
        const __bf16* __restrict__ hr = hbuf + (size_t)(t & 1) * (BSZ * HSZ);
        __bf16* __restrict__       hw = hbuf + (size_t)((t + 1) & 1) * (BSZ * HSZ);

        // Hoist x loads: latency hidden under the 32-iteration WMMA loop.
        float xb[8];
#pragma unroll
        for (int r = 0; r < 8; ++r)
            xb[r] = x[(size_t)(w * 16 + half * 8 + r) * TSZ + t];

        v8f aG = {0,0,0,0,0,0,0,0};
        v8f aI = {0,0,0,0,0,0,0,0};
        v8f aF = {0,0,0,0,0,0,0,0};
        v8f aO = {0,0,0,0,0,0,0,0};

        const __bf16* __restrict__ hrow = hr + (size_t)aRow * HSZ + half * 16;

#pragma unroll 2
        for (int kt = 0; kt < KT; ++kt) {
            // A fragment (16x32 bf16): lane M=lane%16; K = half*16+e, e=0..15.
            AFrag af;
            af.h[0] = *reinterpret_cast<const v8bf*>(hrow + kt * 32);
            af.h[1] = *reinterpret_cast<const v8bf*>(hrow + kt * 32 + 8);

            // B fragments from LDS (step-invariant weights).
            v16bf bG = *reinterpret_cast<const v16bf*>(lds_w + (0 * KT + kt) * 512 + lbo);
            v16bf bI = *reinterpret_cast<const v16bf*>(lds_w + (1 * KT + kt) * 512 + lbo);
            v16bf bF = *reinterpret_cast<const v16bf*>(lds_w + (2 * KT + kt) * 512 + lbo);
            v16bf bO = *reinterpret_cast<const v16bf*>(lds_w + (3 * KT + kt) * 512 + lbo);

            aG = __builtin_amdgcn_wmma_f32_16x16x32_bf16(false, af.v, false, bG, (short)0, aG, false, false);
            aI = __builtin_amdgcn_wmma_f32_16x16x32_bf16(false, af.v, false, bI, (short)0, aI, false, false);
            aF = __builtin_amdgcn_wmma_f32_16x16x32_bf16(false, af.v, false, bF, (short)0, aF, false, false);
            aO = __builtin_amdgcn_wmma_f32_16x16x32_bf16(false, af.v, false, bO, (short)0, aO, false, false);
        }

        // Epilogue: D layout -> lane owns (M = half*8 + r, N = lane%16), r = VGPR idx.
        // All activations branch-free on the trans pipe (co-executes with XDL).
#pragma unroll
        for (int r = 0; r < 8; ++r) {
            const int b  = w * 16 + half * 8 + r;
            const float g  = fast_tanh   (aG[r] + xb[r] * wx_g + bb_g);
            const float ii = fast_sigmoid(aI[r] + xb[r] * wx_i + bb_i);
            const float ff = fast_sigmoid(aF[r] + xb[r] * wx_f + bb_f);
            const float oo = fast_sigmoid(aO[r] + xb[r] * wx_o + bb_o);
            creg[r] = g * ii + creg[r] * ff;
            hw[(size_t)b * HSZ + col] = (__bf16)(fast_tanh(creg[r]) * oo);
        }

        // ---- device-wide barrier (all 64 WGs resident, 1 per WGP) ----
        __builtin_amdgcn_fence(__ATOMIC_RELEASE, "agent");   // flush h stores to device scope
        __syncthreads();
        if (threadIdx.x == 0) {
            __hip_atomic_fetch_add(barrier_ctr, 1u, __ATOMIC_RELAXED, __HIP_MEMORY_SCOPE_AGENT);
            const unsigned target = (unsigned)gridDim.x * (unsigned)(t + 1);
            while (__hip_atomic_load(barrier_ctr, __ATOMIC_RELAXED, __HIP_MEMORY_SCOPE_AGENT) < target) {
                __builtin_amdgcn_s_sleep(1);
            }
        }
        __syncthreads();
        __builtin_amdgcn_fence(__ATOMIC_ACQUIRE, "agent");   // invalidate L0 before reading new h
    }
}

// ---------------------------------------------------------------------------
// Final projection: out[b][c] = h @ W_ph + b_p   (128 x 10, tiny)
// Final h is in buffer 0 (T=256 even: step 255 writes buf[(255+1)&1]=0).
// ---------------------------------------------------------------------------
__global__ void lstm_proj(const __bf16* __restrict__ hbuf,
                          const float* __restrict__ Wph,
                          const float* __restrict__ bp,
                          float* __restrict__ out)
{
    int id = blockIdx.x * blockDim.x + threadIdx.x;
    if (id >= BSZ * CSZ) return;
    int b = id / CSZ, cc = id - b * CSZ;
    float acc = bp[cc];
    const __bf16* hrow = hbuf + (size_t)b * HSZ;
    for (int k = 0; k < HSZ; ++k)
        acc += (float)hrow[k] * Wph[(size_t)k * CSZ + cc];
    out[id] = acc;
}

// ---------------------------------------------------------------------------
extern "C" void kernel_launch(void* const* d_in, const int* in_sizes, int n_in,
                              void* d_out, int out_size, void* d_ws, size_t ws_size,
                              hipStream_t stream)
{
    const float* x   = (const float*)d_in[0];
    const float* Wgx = (const float*)d_in[1];
    const float* Wgh = (const float*)d_in[2];
    const float* bg  = (const float*)d_in[3];
    const float* Wix = (const float*)d_in[4];
    const float* Wih = (const float*)d_in[5];
    const float* bi  = (const float*)d_in[6];
    const float* Wfx = (const float*)d_in[7];
    const float* Wfh = (const float*)d_in[8];
    const float* bf  = (const float*)d_in[9];
    const float* Wox = (const float*)d_in[10];
    const float* Woh = (const float*)d_in[11];
    const float* bo  = (const float*)d_in[12];
    const float* Wph = (const float*)d_in[13];
    const float* bp  = (const float*)d_in[14];
    float* out = (float*)d_out;

    // Workspace layout:
    //   [0, 8MB)           packed bf16 gate weights (4 * 1024 * 1024)
    //   [8MB, 8MB+512KB)   h double buffer (2 * 128*1024 bf16)
    //   [8MB+512KB, +4)    barrier counter
    char* ws = (char*)d_ws;
    __bf16*  packedW = (__bf16*)ws;
    __bf16*  hbuf    = (__bf16*)(ws + 8ull * 1024 * 1024);
    unsigned* ctr    = (unsigned*)(ws + 8ull * 1024 * 1024 + 512ull * 1024);

    // Raise dynamic-LDS cap to 128 KB (host-side attr; enqueues nothing, so
    // it is graph-capture safe and deterministic). Ignore error if already set.
    (void)hipFuncSetAttribute((const void*)lstm_persistent,
                              hipFuncAttributeMaxDynamicSharedMemorySize,
                              128 * 1024);

    // Reset h (h0 = 0) and barrier counter every call (graph-capture safe).
    hipMemsetAsync(hbuf, 0, 512 * 1024 + 256, stream);

    // Repack weights every call (deterministic): 4M elements.
    lstm_pack_w<<<dim3((4u * NWG * KT * 32 * 16) / 256), dim3(256), 0, stream>>>(
        Wgh, Wih, Wfh, Woh, packedW);

    lstm_persistent<<<dim3(NWG), dim3(256), 128 * 1024, stream>>>(
        x, Wgx, bg, Wix, bi, Wfx, bf, Wox, bo, packedW, hbuf, ctr);

    lstm_proj<<<dim3((BSZ * CSZ + 255) / 256), dim3(256), 0, stream>>>(
        hbuf, Wph, bp, out);
}
